// RegionProposalNetwork_29540785062165
// MI455X (gfx1250) — matrix operand order repaired and checked by reading, not compiled, not run on hardware
//
#include <hip/hip_runtime.h>
#include <cstdint>
#include <cmath>

typedef __attribute__((ext_vector_type(16))) _Float16 v16h;
typedef __attribute__((ext_vector_type(8)))  float    v8f;
typedef int vi4 __attribute__((vector_size(16)));

#define AS1 __attribute__((address_space(1)))
#define AS3 __attribute__((address_space(3)))

#if defined(__has_builtin)
#if __has_builtin(__builtin_amdgcn_global_load_async_to_lds_b128) && \
    __has_builtin(__builtin_amdgcn_s_wait_asynccnt)
#define USE_ASYNC 1
#endif
#endif
#ifndef USE_ASYNC
#define USE_ASYNC 0
#endif

constexpr int NB   = 4;
constexpr int HH   = 64;
constexpr int WW   = 64;
constexpr int NC   = 512;
constexpr int KTOT = 9 * 512;          // 4608
constexpr int NIT  = KTOT / 32;        // 144 K-steps
constexpr int NAN_ = HH * WW * 9;      // 36864 anchors per image
constexpr int PRE  = 12000;
constexpr int POST = 600;

// ---------------------------------------------------------------- converts
__global__ void k_cvt_w(const float* __restrict__ w, _Float16* __restrict__ wt) {
  int idx = blockIdx.x * 256 + threadIdx.x;            // over NC*KTOT, [n][k]
  if (idx >= NC * KTOT) return;
  int n = idx / KTOT, k = idx - n * KTOT;
  int kpos = k >> 9, c = k & 511;
  int ky = kpos / 3, kx = kpos - 3 * ky;
  wt[idx] = (_Float16)w[((n * NC + c) * 3 + ky) * 3 + kx];   // OIHW -> [n][k]
}

__global__ void k_cvt_x(const float* __restrict__ x, _Float16* __restrict__ xh) {
  int idx = blockIdx.x * 256 + threadIdx.x;            // dest NHWC index
  if (idx >= NB * HH * WW * NC) return;
  int c = idx & 511;
  int xx = (idx >> 9) & 63;
  int yy = (idx >> 15) & 63;
  int b  = idx >> 21;
  xh[idx] = (_Float16)x[((b * NC + c) * HH + yy) * WW + xx];
}

// --------------------------------------------------- 3x3 conv implicit GEMM
// Block tile 128x64, BK=32, 8 waves, 32x32 per wave (4 wmma / K-step / wave).
// Double-buffered LDS; async global->LDS staging when available.
__global__ __launch_bounds__(256) void k_conv(const _Float16* __restrict__ xh,
                                              const _Float16* __restrict__ wt,
                                              const float* __restrict__ bias,
                                              _Float16* __restrict__ feat) {
  __shared__ __align__(16) _Float16 As[2][128][40];   // [m][k], padded stride
  __shared__ __align__(16) _Float16 Bs[2][64][40];    // [n][k], padded stride

  const int t    = threadIdx.x;
  const int m0   = blockIdx.x * 128;
  const int n0   = blockIdx.y * 64;
  const int lane = t & 31, wave = t >> 5;
  const int wm   = wave & 3, wn = wave >> 2;
  const int hi   = lane >> 4, ln = lane & 15;

  v8f acc00 = {}, acc01 = {}, acc10 = {}, acc11 = {};

  // cooperative-load roles: A = 128 rows x 4 segs (2 tasks/thread),
  //                         B =  64 rows x 4 segs (1 task/thread)
  const int seg = t & 3;
  const int rA0 = t >> 2;        // 0..63
  const int rA1 = rA0 + 64;      // 64..127
  const int rB  = t >> 2;        // 0..63

  const int mA0 = m0 + rA0, mA1 = m0 + rA1;
  const int bb0 = mA0 >> 12, yy0 = (mA0 >> 6) & 63, xx0 = mA0 & 63;
  const int bb1 = mA1 >> 12, yy1 = (mA1 >> 6) & 63, xx1 = mA1 & 63;

  auto issue = [&](int it, int buf) {
    const int k0   = it * 32;
    const int kpos = k0 >> 9;
    const int dy   = kpos / 3 - 1;
    const int dx   = (kpos - 3 * (kpos / 3)) - 1;
    const int c    = (k0 & 511) + seg * 8;
    // ---- A row 0
    {
      const int iy = yy0 + dy, ix = xx0 + dx;
      _Float16* dst = &As[buf][rA0][seg * 8];
      if ((unsigned)iy < 64u && (unsigned)ix < 64u) {
        const _Float16* src = &xh[(size_t)((bb0 * 64 + iy) * 64 + ix) * 512 + c];
#if USE_ASYNC
        __builtin_amdgcn_global_load_async_to_lds_b128(
            (AS1 vi4*)src, (AS3 vi4*)dst, 0, 0);
#else
        *(float4*)dst = *(const float4*)src;
#endif
      } else {
        *(float4*)dst = make_float4(0.f, 0.f, 0.f, 0.f);
      }
    }
    // ---- A row 1
    {
      const int iy = yy1 + dy, ix = xx1 + dx;
      _Float16* dst = &As[buf][rA1][seg * 8];
      if ((unsigned)iy < 64u && (unsigned)ix < 64u) {
        const _Float16* src = &xh[(size_t)((bb1 * 64 + iy) * 64 + ix) * 512 + c];
#if USE_ASYNC
        __builtin_amdgcn_global_load_async_to_lds_b128(
            (AS1 vi4*)src, (AS3 vi4*)dst, 0, 0);
#else
        *(float4*)dst = *(const float4*)src;
#endif
      } else {
        *(float4*)dst = make_float4(0.f, 0.f, 0.f, 0.f);
      }
    }
    // ---- B
    {
      const _Float16* src = &wt[(size_t)(n0 + rB) * KTOT + k0 + seg * 8];
      _Float16* dst = &Bs[buf][rB][seg * 8];
#if USE_ASYNC
      __builtin_amdgcn_global_load_async_to_lds_b128(
          (AS1 vi4*)src, (AS3 vi4*)dst, 0, 0);
#else
      *(float4*)dst = *(const float4*)src;
#endif
    }
  };

  issue(0, 0);

  const int ar0 = wm * 32 + ln;
  const int bn0 = wn * 32 + ln;

  for (int it = 0; it < NIT; ++it) {
    const int cur = it & 1;
#if USE_ASYNC
    __builtin_amdgcn_s_wait_asynccnt(0);
#endif
    __syncthreads();
    if (it + 1 < NIT) issue(it + 1, cur ^ 1);

    // fragments (ISA 7.12.2 layouts) -- each pair of 16B LDS loads
    union U { v16h h; uint4 q[2]; };
    U af0, af1, bf0, bf1;
    af0.q[0] = *(const uint4*)&As[cur][ar0][hi * 8];
    af0.q[1] = *(const uint4*)&As[cur][ar0][hi * 8 + 16];
    af1.q[0] = *(const uint4*)&As[cur][ar0 + 16][hi * 8];
    af1.q[1] = *(const uint4*)&As[cur][ar0 + 16][hi * 8 + 16];
    bf0.q[0] = *(const uint4*)&Bs[cur][bn0][hi * 8];
    bf0.q[1] = *(const uint4*)&Bs[cur][bn0][hi * 8 + 16];
    bf1.q[0] = *(const uint4*)&Bs[cur][bn0 + 16][hi * 8];
    bf1.q[1] = *(const uint4*)&Bs[cur][bn0 + 16][hi * 8 + 16];

    acc00 = __builtin_amdgcn_wmma_f32_16x16x32_f16(false, af0.h, false, bf0.h,
                                                   (short)0, acc00, false, false);
    acc01 = __builtin_amdgcn_wmma_f32_16x16x32_f16(false, af0.h, false, bf1.h,
                                                   (short)0, acc01, false, false);
    acc10 = __builtin_amdgcn_wmma_f32_16x16x32_f16(false, af1.h, false, bf0.h,
                                                   (short)0, acc10, false, false);
    acc11 = __builtin_amdgcn_wmma_f32_16x16x32_f16(false, af1.h, false, bf1.h,
                                                   (short)0, acc11, false, false);
  }

  // epilogue: bias + relu, store f16 NHWC feature map
  const int mb = m0 + wm * 32 + hi * 8;     // + r (0..7), +16 for second M sub
  const int nb = n0 + wn * 32 + ln;         // +16 for second N sub
  const float bs0 = bias[nb];
  const float bs1 = bias[nb + 16];
#pragma unroll
  for (int r = 0; r < 8; ++r) {
    float v00 = acc00[r] + bs0; v00 = v00 > 0.f ? v00 : 0.f;
    float v01 = acc01[r] + bs1; v01 = v01 > 0.f ? v01 : 0.f;
    float v10 = acc10[r] + bs0; v10 = v10 > 0.f ? v10 : 0.f;
    float v11 = acc11[r] + bs1; v11 = v11 > 0.f ? v11 : 0.f;
    feat[(size_t)(mb + r) * 512 + nb]            = (_Float16)v00;
    feat[(size_t)(mb + r) * 512 + nb + 16]       = (_Float16)v01;
    feat[(size_t)(mb + 16 + r) * 512 + nb]       = (_Float16)v10;
    feat[(size_t)(mb + 16 + r) * 512 + nb + 16]  = (_Float16)v11;
  }
}

// ----------------------------------------------- 1x1 heads + 2-way softmax
__global__ __launch_bounds__(64) void k_heads(const _Float16* __restrict__ feat,
                                              const float* __restrict__ lw,
                                              const float* __restrict__ lb,
                                              const float* __restrict__ sw,
                                              const float* __restrict__ sb,
                                              float* __restrict__ out_locs,
                                              float* __restrict__ out_scores,
                                              float* __restrict__ probs) {
  __shared__ _Float16 f[512];
  __shared__ float scsh[18];
  const int pos = blockIdx.x;                // b*4096 + hw
  const int t = threadIdx.x;
  for (int i = t; i < 512; i += 64) f[i] = feat[(size_t)pos * 512 + i];
  __syncthreads();
  const int b = pos >> 12, hw = pos & 4095;
  if (t < 54) {
    const float* wrow;
    float acc;
    int o;
    if (t < 36) { o = t;      wrow = lw + o * 512; acc = lb[o]; }
    else        { o = t - 36; wrow = sw + o * 512; acc = sb[o]; }
    for (int c = 0; c < 512; ++c) acc += (float)f[c] * wrow[c];
    if (t < 36) {
      out_locs[((size_t)b * NAN_ + hw * 9 + (o >> 2)) * 4 + (o & 3)] = acc;
    } else {
      out_scores[((size_t)b * NAN_ + hw * 9 + (o >> 1)) * 2 + (o & 1)] = acc;
      scsh[o] = acc;
    }
  }
  __syncthreads();
  if (t < 9) {
    float s0 = scsh[2 * t], s1 = scsh[2 * t + 1];
    float mx = fmaxf(s0, s1);
    float e0 = __expf(s0 - mx), e1 = __expf(s1 - mx);
    probs[(size_t)b * NAN_ + hw * 9 + t] = e1 / (e0 + e1);
  }
}

// -------------------------------- anchors + loc2bbox + clip + minsize mask
__global__ void k_bbox(const float* __restrict__ locs, const float* __restrict__ probs,
                       float* __restrict__ bbox, float* __restrict__ sc,
                       float* __restrict__ out_anch, float* __restrict__ out_ridx,
                       const int* __restrict__ ihp, const int* __restrict__ iwp) {
  int idx = blockIdx.x * 256 + threadIdx.x;
  if (idx >= NB * NAN_) return;
  int b = idx / NAN_, i = idx - b * NAN_;
  int a = i % 9, hw = i / 9;
  int y = hw >> 6, x = hw & 63;
  const float rat[3] = {0.5f, 1.f, 2.f};
  const float scl[3] = {8.f, 16.f, 32.f};
  float r = rat[a / 3], s = scl[a % 3];
  float ah = 16.f * s * sqrtf(r);
  float aw = 16.f * s * sqrtf(1.f / r);
  float sx = x * 16.f, sy = y * 16.f;
  float a0 = sx - 0.5f * ah, a1 = sy - 0.5f * aw;
  float a2 = sx + 0.5f * ah, a3 = sy + 0.5f * aw;
  if (b == 0) {
    out_anch[i * 4 + 0] = a0; out_anch[i * 4 + 1] = a1;
    out_anch[i * 4 + 2] = a2; out_anch[i * 4 + 3] = a3;
  }
  float src_w = a2 - a0, src_h = a3 - a1;
  float ctr_x = a0 + 0.5f * src_w, ctr_y = a1 + 0.5f * src_h;
  const float* L = locs + (size_t)idx * 4;
  float cx = L[0] * src_w + ctr_x, cy = L[1] * src_h + ctr_y;
  float bw = __expf(L[2]) * src_w, bh = __expf(L[3]) * src_h;
  float imh = (float)ihp[0], imw = (float)iwp[0];
  float y1 = fminf(fmaxf(cy - 0.5f * bh, 0.f), imh);
  float x1 = fminf(fmaxf(cx - 0.5f * bw, 0.f), imw);
  float y2 = fminf(fmaxf(cy + 0.5f * bh, 0.f), imh);
  float x2 = fminf(fmaxf(cx + 0.5f * bw, 0.f), imw);
  bool keep = (y2 - y1 >= 16.f) && (x2 - x1 >= 16.f);
  float p = probs[idx];
  float* bo = bbox + (size_t)idx * 4;
  bo[0] = y1; bo[1] = x1; bo[2] = y2; bo[3] = x2;
  sc[idx] = keep ? p : -INFINITY;
  if (i < POST) out_ridx[b * POST + i] = (float)b;
}

// -------------------- exact stable top-k rank (descending, tie = lower idx)
__global__ __launch_bounds__(256) void k_rank(const float* __restrict__ sc,
                                              int* __restrict__ rank) {
  __shared__ float ch[4096];
  const int b = blockIdx.y;
  const int i = blockIdx.x * 256 + threadIdx.x;      // exact: 144*256 == NAN_
  const float* S = sc + (size_t)b * NAN_;
  const float si = S[i];
  int cnt = 0;
  for (int c0 = 0; c0 < NAN_; c0 += 4096) {
    __syncthreads();
    for (int j = threadIdx.x; j < 4096; j += 256) ch[j] = S[c0 + j];
    __syncthreads();
#pragma unroll 4
    for (int j = 0; j < 4096; ++j) {
      float sj = ch[j];
      int gj = c0 + j;
      cnt += (sj > si || (sj == si && gj < i)) ? 1 : 0;
    }
  }
  rank[(size_t)b * NAN_ + i] = cnt;
}

__global__ void k_scatter(const float* __restrict__ bbox, const float* __restrict__ sc,
                          const int* __restrict__ rank, float* __restrict__ obox,
                          float* __restrict__ osc) {
  int idx = blockIdx.x * 256 + threadIdx.x;
  if (idx >= NB * NAN_) return;
  int b = idx / NAN_;
  int r = rank[idx];
  if (r < PRE) {
    const float* src = bbox + (size_t)idx * 4;
    float* dst = obox + ((size_t)b * PRE + r) * 4;
    dst[0] = src[0]; dst[1] = src[1]; dst[2] = src[2]; dst[3] = src[3];
    osc[(size_t)b * PRE + r] = sc[idx];
  }
}

// ------------------------------------------------------ sequential NMS(600)
__global__ __launch_bounds__(1024) void k_nms(const float* __restrict__ obox,
                                              const float* __restrict__ osc,
                                              float* __restrict__ rois) {
  __shared__ unsigned char sup[PRE];
  __shared__ int s_pick, s_last;
  __shared__ float4 s_box;
  __shared__ float s_area;
  const int b = blockIdx.x, t = threadIdx.x;
  const float4* BX = (const float4*)(obox + (size_t)b * PRE * 4);
  float4 bx[12]; float ar[12];
#pragma unroll
  for (int j = 0; j < 12; ++j) {
    int e = j * 1024 + t;
    if (e < PRE) {
      float4 v = BX[e];
      bx[j] = v;
      ar[j] = (v.z - v.x) * (v.w - v.y);
      sup[e] = (osc[(size_t)b * PRE + e] <= -1e30f) ? 1 : 0;
    }
  }
  if (t == 0) s_last = 0;
  __syncthreads();
  for (int it = 0; it < POST; ++it) {
    if (t == 0) s_pick = 0x7fffffff;
    __syncthreads();
    int lm = 0x7fffffff;
#pragma unroll
    for (int j = 0; j < 12; ++j) {
      int e = j * 1024 + t;
      if (e < PRE && !sup[e]) { lm = min(lm, e); }
    }
    for (int off = 16; off > 0; off >>= 1)
      lm = min(lm, __shfl_down(lm, off, 32));          // wave32 reduce
    if ((t & 31) == 0) atomicMin(&s_pick, lm);
    __syncthreads();
    if (t == 0) {
      int p = s_pick;
      if (p == 0x7fffffff) p = s_last; else s_last = p;
      sup[p] = 1;
      float4 pb = BX[p];
      s_box = pb;
      s_area = (pb.z - pb.x) * (pb.w - pb.y);
      float* rr = rois + ((size_t)b * POST + it) * 4;
      rr[0] = pb.x; rr[1] = pb.y; rr[2] = pb.z; rr[3] = pb.w;
    }
    __syncthreads();
    float4 pb = s_box; float pa = s_area;
#pragma unroll
    for (int j = 0; j < 12; ++j) {
      int e = j * 1024 + t;
      if (e < PRE && !sup[e]) {
        float yy1 = fmaxf(pb.x, bx[j].x);
        float xx1 = fmaxf(pb.y, bx[j].y);
        float yy2 = fminf(pb.z, bx[j].z);
        float xx2 = fminf(pb.w, bx[j].w);
        float inter = fmaxf(yy2 - yy1, 0.f) * fmaxf(xx2 - xx1, 0.f);
        float iou = inter / fmaxf(pa + ar[j] - inter, 1e-9f);
        if (iou > 0.7f) sup[e] = 1;
      }
    }
    __syncthreads();
  }
}

// ------------------------------------------------------------------ launch
extern "C" void kernel_launch(void* const* d_in, const int* in_sizes, int n_in,
                              void* d_out, int out_size, void* d_ws, size_t ws_size,
                              hipStream_t stream) {
  const float* x       = (const float*)d_in[0];
  const float* conv_w  = (const float*)d_in[1];
  const float* conv_b  = (const float*)d_in[2];
  const float* score_w = (const float*)d_in[3];
  const float* score_b = (const float*)d_in[4];
  const float* loc_w   = (const float*)d_in[5];
  const float* loc_b   = (const float*)d_in[6];
  const int*   img_h   = (const int*)d_in[7];
  const int*   img_w   = (const int*)d_in[8];

  float* out        = (float*)d_out;
  float* out_locs   = out;                  // 4*36864*4
  float* out_scores = out + 589824;         // 4*36864*2
  float* out_rois   = out + 884736;         // 4*600*4
  float* out_ridx   = out + 894336;         // 4*600
  float* out_anch   = out + 896736;         // 36864*4

  char* ws = (char*)d_ws;
  size_t off = 0;
  auto alloc = [&](size_t bytes) -> char* {
    char* p = ws + off;
    off = (off + bytes + 255) & ~(size_t)255;
    return p;
  };
  _Float16* Wt   = (_Float16*)alloc((size_t)NC * KTOT * 2);
  _Float16* Xh   = (_Float16*)alloc((size_t)NB * HH * WW * NC * 2);
  _Float16* Ft   = (_Float16*)alloc((size_t)NB * HH * WW * NC * 2);
  float* probs   = (float*)alloc((size_t)NB * NAN_ * 4);
  float* bbox    = (float*)alloc((size_t)NB * NAN_ * 16);
  float* scb     = (float*)alloc((size_t)NB * NAN_ * 4);
  int*   rankb   = (int*)alloc((size_t)NB * NAN_ * 4);
  float* obox    = (float*)alloc((size_t)NB * PRE * 16);
  float* oscb    = (float*)alloc((size_t)NB * PRE * 4);
  (void)ws_size; (void)in_sizes; (void)n_in; (void)out_size;

  hipLaunchKernelGGL(k_cvt_w, dim3((NC * KTOT + 255) / 256), dim3(256), 0, stream,
                     conv_w, Wt);
  hipLaunchKernelGGL(k_cvt_x, dim3((NB * HH * WW * NC + 255) / 256), dim3(256), 0,
                     stream, x, Xh);
  hipLaunchKernelGGL(k_conv, dim3(128, 8), dim3(256), 0, stream, Xh, Wt, conv_b, Ft);
  hipLaunchKernelGGL(k_heads, dim3(NB * HH * WW), dim3(64), 0, stream, Ft, loc_w,
                     loc_b, score_w, score_b, out_locs, out_scores, probs);
  hipLaunchKernelGGL(k_bbox, dim3((NB * NAN_ + 255) / 256), dim3(256), 0, stream,
                     out_locs, probs, bbox, scb, out_anch, out_ridx, img_h, img_w);
  hipLaunchKernelGGL(k_rank, dim3(NAN_ / 256, NB), dim3(256), 0, stream, scb, rankb);
  hipLaunchKernelGGL(k_scatter, dim3((NB * NAN_ + 255) / 256), dim3(256), 0, stream,
                     bbox, scb, rankb, obox, oscb);
  hipLaunchKernelGGL(k_nms, dim3(NB), dim3(1024), 0, stream, obox, oscb, out_rois);
}
